// criterionHinge_65412351918333
// MI455X (gfx1250) — matrix operand order repaired and checked by reading, not compiled, not run on hardware
//
#include <hip/hip_runtime.h>
#include <hip/hip_bf16.h>

typedef _Float16 v16h __attribute__((ext_vector_type(16)));
typedef _Float16 v8h  __attribute__((ext_vector_type(8)));
typedef float    v8f  __attribute__((ext_vector_type(8)));

#define N_ROWS   2048
#define T_TIMES  16
#define DF       64
#define TILES    (N_ROWS / 16)                  // 128
#define NTILE    (TILES * (TILES + 1) / 2)      // 8256 upper-tri tiles incl. diag
#define XNH_BYTES ((size_t)N_ROWS * DF * 2)     // 256 KB f16 normalized X

// ---------------------------------------------------------------------------
// Kernel 1: row-normalize X (f32) -> xn (f16), wave per row.
// ---------------------------------------------------------------------------
__global__ __launch_bounds__(256) void fair_normalize(
    const float* __restrict__ X, _Float16* __restrict__ xnh) {
  const int wave = threadIdx.x >> 5;
  const int lane = threadIdx.x & 31;
  const int row  = blockIdx.x * 8 + wave;

  const float* xr = X + (size_t)row * DF;
  float2 v = *(const float2*)(xr + 2 * lane);   // 64 feats / 32 lanes
  float ss = v.x * v.x + v.y * v.y;
#pragma unroll
  for (int off = 16; off > 0; off >>= 1) ss += __shfl_xor(ss, off, 32);
  const float inv = 1.0f / fmaxf(sqrtf(ss), 1e-12f);

  _Float16* o = xnh + (size_t)row * DF + 2 * lane;
  o[0] = (_Float16)(v.x * inv);
  o[1] = (_Float16)(v.y * inv);
}

// ---------------------------------------------------------------------------
// Kernel 2: one wave per 16x16 tile of the upper triangle.
//   WMMA gram tile (2 x v_wmma_f32_16x16x32_f16, K=64) fused with the
//   16-times hinge accumulation. Writes one partial per tile.
// ---------------------------------------------------------------------------
__global__ __launch_bounds__(256) void fair_tile(
    const _Float16* __restrict__ xnh, const float* __restrict__ pred,
    const float* __restrict__ scale_p, float* __restrict__ partials) {
  const int lane = threadIdx.x & 31;
  const int wave = threadIdx.x >> 5;
  const int t    = blockIdx.x * 8 + wave;       // tile id in [0, NTILE)

  // Map flat tile id -> (tj, tk) with tk >= tj (row-major upper triangle).
  const int M = TILES;
  int r = (int)(((float)(2 * M + 1) -
                 sqrtf((float)(2 * M + 1) * (float)(2 * M + 1) - 8.0f * (float)t)) * 0.5f);
  if (r < 0) r = 0;
  if (r > M - 1) r = M - 1;
  while (r > 0 && t < r * M - (r * (r - 1)) / 2) --r;
  while (t >= (r + 1) * M - ((r + 1) * r) / 2) ++r;
  const int sr = r * M - (r * (r - 1)) / 2;
  const int tj = r;
  const int tk = r + (t - sr);
  const int j0 = tj * 16, k0 = tk * 16;

  const int mrow = lane & 15;                   // per-lane M (for A) / N (for B)
  const int hi   = lane >> 4;                   // half-wave select

  // ---- gram tile via WMMA: c[e] = dot(xn[j0+M_e], xn[k0+N]) ----
  v8f c = {};
#pragma unroll
  for (int kc = 0; kc < DF; kc += 32) {
    // A (16x32 f16): lane<16 -> K = kc+0..7 & kc+16..23 ; lane>=16 -> +8
    const _Float16* ar = xnh + (size_t)(j0 + mrow) * DF + kc + hi * 8;
    v16h a;
    ((v8h*)&a)[0] = *(const v8h*)(ar);
    ((v8h*)&a)[1] = *(const v8h*)(ar + 16);
    // B (32x16 f16), B[kk][n] = xn[k0+n][kk]: same K order as A
    const _Float16* br = xnh + (size_t)(k0 + mrow) * DF + kc + hi * 8;
    v16h b;
    ((v8h*)&b)[0] = *(const v8h*)(br);
    ((v8h*)&b)[1] = *(const v8h*)(br + 16);
    c = __builtin_amdgcn_wmma_f32_16x16x32_f16(false, a, false, b,
                                               (short)0, c, false, false);
  }

  const float scale_v = scale_p[0];
  const int   kglob   = k0 + mrow;              // this lane's global column k

  // pred[k, 0..15] held in registers (per-lane, shared across all 8 rows)
  float pkf[T_TIMES];
  {
    const float4* pkp = (const float4*)(pred + (size_t)kglob * T_TIMES);
#pragma unroll
    for (int q = 0; q < 4; ++q) {
      float4 v = pkp[q];
      pkf[4 * q + 0] = v.x; pkf[4 * q + 1] = v.y;
      pkf[4 * q + 2] = v.z; pkf[4 * q + 3] = v.w;
    }
  }

  float wsum = 0.0f;
#pragma unroll
  for (int e = 0; e < 8; ++e) {
    const int   jglob = j0 + e + hi * 8;        // C layout: VGPR e -> M = e (+8)
    const float thr   = scale_v * (1.0f - c[e]);
    // pred[j, 0..15]: address uniform per half-wave -> broadcast from L0
    const float4* pjp = (const float4*)(pred + (size_t)jglob * T_TIMES);
    float es = 0.0f;
#pragma unroll
    for (int q = 0; q < 4; ++q) {
      float4 v = pjp[q];
      es += fmaxf(fabsf(v.x - pkf[4 * q + 0]) - thr, 0.0f);
      es += fmaxf(fabsf(v.y - pkf[4 * q + 1]) - thr, 0.0f);
      es += fmaxf(fabsf(v.z - pkf[4 * q + 2]) - thr, 0.0f);
      es += fmaxf(fabsf(v.w - pkf[4 * q + 3]) - thr, 0.0f);
    }
    wsum += (kglob > jglob) ? es : 0.0f;        // strict upper triangle only
  }

#pragma unroll
  for (int off = 16; off > 0; off >>= 1) wsum += __shfl_xor(wsum, off, 32);
  if (lane == 0) partials[t] = wsum;            // each slot written exactly once
}

// ---------------------------------------------------------------------------
// Kernel 3: deterministic fixed-order reduction + finalize.
// ---------------------------------------------------------------------------
__global__ __launch_bounds__(256) void fair_finalize(
    const float* __restrict__ partials, int nparts,
    const float* __restrict__ target_p, float* __restrict__ out) {
  __shared__ float sm[256];
  float s = 0.0f;
  for (int i = threadIdx.x; i < nparts; i += 256) s += partials[i];
  sm[threadIdx.x] = s;
  __syncthreads();
  for (int off = 128; off > 0; off >>= 1) {
    if ((int)threadIdx.x < off) sm[threadIdx.x] += sm[threadIdx.x + off];
    __syncthreads();
  }
  if (threadIdx.x == 0) {
    const float denom = (float)N_ROWS * (float)(N_ROWS - 1) * (float)T_TIMES;
    const float mf = 2.0f * sm[0] / denom;
    out[0] = fmaxf(0.0f, mf - target_p[0]);
  }
}

// ---------------------------------------------------------------------------
extern "C" void kernel_launch(void* const* d_in, const int* in_sizes, int n_in,
                              void* d_out, int out_size, void* d_ws, size_t ws_size,
                              hipStream_t stream) {
  (void)in_sizes; (void)n_in; (void)out_size; (void)ws_size;

  const float* target = (const float*)d_in[0];   // scalar
  const float* pred   = (const float*)d_in[1];   // [2048,16] f32
  const float* X      = (const float*)d_in[2];   // [2048,64] f32
  const float* scale  = (const float*)d_in[3];   // scalar
  // d_in[4] = ntimes (int) — compile-time constant T_TIMES here

  char*     ws       = (char*)d_ws;
  _Float16* xnh      = (_Float16*)ws;            // 256 KB normalized f16 X
  float*    partials = (float*)(ws + XNH_BYTES); // 8256 floats

  fair_normalize<<<N_ROWS / 8, 256, 0, stream>>>(X, xnh);
  fair_tile<<<NTILE / 8, 256, 0, stream>>>(xnh, pred, scale, partials);
  fair_finalize<<<1, 256, 0, stream>>>(partials, NTILE, target, (float*)d_out);
}
